// ImageDFRSystem_37108517437480
// MI455X (gfx1250) — matrix operand order repaired and checked by reading, not compiled, not run on hardware
//
#include <hip/hip_runtime.h>
#include <hip/hip_bf16.h>

typedef _Float16 v16h __attribute__((ext_vector_type(16)));
typedef float    v8f  __attribute__((ext_vector_type(8)));

#if __has_builtin(__builtin_amdgcn_tanhf)
#define DEV_TANH(x) __builtin_amdgcn_tanhf(x)
#else
#define DEV_TANH(x) tanhf(x)
#endif

#define NUM_DFR 6
#define NHID 10
#define STEPS 49
#define KIN 196
#define FEEDBACK 0.5f

// ---------------------------------------------------------------------------
// Pre-pack kernel (1 wave): build f16 WMMA fragments in workspace.
//  aFrag[6][32][16]  : A operands (16x32 f16) for the 6 cells (W0 col / Wrest, zero padded)
//  bFrag[28][32][16] : B operands (32x16 f16) of W_p1^T, frag id = nt*7 + kc
// A layout: lane L holds row M=L%16; halves j -> K = j (lo lanes) / 16+j (hi lanes)
// B layout: lane L holds col N=L%16; halves j -> K = j (lo lanes) / 16+j (hi lanes)
// ---------------------------------------------------------------------------
__global__ void pack_weights_kernel(const float* __restrict__ W_p1,   // [49,196]
                                    const float* __restrict__ W0,     // [10,1]
                                    const float* __restrict__ Wrest,  // [5,10,10]
                                    _Float16* __restrict__ aFrag,
                                    _Float16* __restrict__ bFrag)
{
    int lane = threadIdx.x & 31;
    int row  = lane & 15;
    int hi   = lane >> 4;

    // A fragments for the 6 cells
    for (int i = 0; i < NUM_DFR; ++i) {
        for (int j = 0; j < 16; ++j) {
            int k  = j + (hi ? 16 : 0);       // lo lanes: {0..15}, hi lanes: {16..31}
            float v = 0.0f;
            if (i == 0) {
                if (row < NHID && k == 0) v = W0[row];
            } else {
                if (row < NHID && k < NHID) v = Wrest[(size_t)(i - 1) * 100 + row * 10 + k];
            }
            aFrag[((size_t)i * 32 + lane) * 16 + j] = (_Float16)v;
        }
    }
    // B fragments of W_p1^T : B[k][n] = W_p1[n][k]
    for (int nt = 0; nt < 4; ++nt) {
        for (int kc = 0; kc < 7; ++kc) {
            int n = nt * 16 + row;
            for (int j = 0; j < 16; ++j) {
                int k = kc * 32 + hi * 16 + j;
                float v = (n < STEPS && k < KIN) ? W_p1[(size_t)n * KIN + k] : 0.0f;
                bFrag[((size_t)(nt * 7 + kc) * 32 + lane) * 16 + j] = (_Float16)v;
            }
        }
    }
}

// ---------------------------------------------------------------------------
// helpers for the recurrence (state h: v8f in D layout:
//   lanes 0-15: batch n=lane, hidden m = r      (r = vector index 0..7)
//   lanes16-31: batch n=lane-16, hidden m = r+8 (only r=0,1 live; rest stay 0)
// ---------------------------------------------------------------------------
__device__ __forceinline__ v8f roll_feedback(v8f h, int hi)
{
    // new[m] = FEEDBACK * old[(m-1) mod 10]
    float m9 = __shfl_xor(h[1], 16, 32);  // lo lanes receive hidden 9 from hi half
    float m7 = __shfl_xor(h[7], 16, 32);  // hi lanes receive hidden 7 from lo half
    v8f c;
    c[0] = FEEDBACK * (hi ? m7 : m9);     // lo: m0<-m9 ; hi: m8<-m7
    c[1] = FEEDBACK * h[0];               // lo: m1<-m0 ; hi: m9<-m8
#pragma unroll
    for (int r = 2; r < 8; ++r)
        c[r] = hi ? 0.0f : FEEDBACK * h[r - 1]; // hi dead slots m10..15 stay 0
    return c;
}

__device__ __forceinline__ v16h pack_state_B(v8f hp, int hi)
{
    // B (32x16): lo lanes carry K=0..15 (hidden 0..9 live), hi lanes K=16..31 -> zero
    float h8 = __shfl_xor(hp[0], 16, 32);
    float h9 = __shfl_xor(hp[1], 16, 32);
    v16h b;
#pragma unroll
    for (int j = 0; j < 16; ++j) b[j] = (_Float16)0.0f;
    if (!hi) {
#pragma unroll
        for (int j = 0; j < 8; ++j) b[j] = (_Float16)hp[j];
        b[8] = (_Float16)h8;
        b[9] = (_Float16)h9;
    }
    return b;
}

// ---------------------------------------------------------------------------
// Fused main kernel: fc_p1 GEMM (WMMA) -> 49-step 6-cell DFR scan (WMMA) -> head
// One wave owns a 16-batch tile. 4 waves / block.
// ---------------------------------------------------------------------------
__global__ __launch_bounds__(128) void dfr_fused_kernel(
    const float* __restrict__ x,         // [B,196]
    const float* __restrict__ cell_out,  // [6,B,10]
    const float* __restrict__ W1,        // [8,10]
    const float* __restrict__ W2,        // [10,8]
    const float* __restrict__ b2,        // [10]
    const _Float16* __restrict__ aFrag,  // [6][32][16]
    const _Float16* __restrict__ bFrag,  // [28][32][16]
    float* __restrict__ out,             // [B,10]
    int Btot)
{
    __shared__ float ldsXp[4][16][52];   // per-wave pixel sequences (stride 52: conflict-free)
    __shared__ float ldsW1[80];
    __shared__ float ldsW2[80];
    __shared__ float ldsB2[16];

    int tid = threadIdx.x;
    if (tid < 80) { ldsW1[tid] = W1[tid]; ldsW2[tid] = W2[tid]; }
    if (tid < 10) ldsB2[tid] = b2[tid];

    int wv   = tid >> 5;
    int lane = tid & 31;
    int row  = lane & 15;
    int hi   = lane >> 4;
    int base = (blockIdx.x * 4 + wv) * 16;   // 16 batch rows per wave

    // ---------------- Phase 1: xp tile = x[base:base+16] @ W_p1^T (WMMA) ----
    v8f acc[4];
#pragma unroll
    for (int nt = 0; nt < 4; ++nt)
#pragma unroll
        for (int r = 0; r < 8; ++r) acc[nt][r] = 0.0f;

    const float* xr = x + (size_t)(base + row) * KIN;
    for (int kc = 0; kc < 7; ++kc) {
        // A fragment from x (f32 -> f16 on the fly, zero-pad K 196..223)
        v16h a;
#pragma unroll
        for (int j = 0; j < 16; ++j) {
            int k  = kc * 32 + hi * 16 + j;
            a[j] = (k < KIN) ? (_Float16)xr[k] : (_Float16)0.0f;
        }
#pragma unroll
        for (int nt = 0; nt < 4; ++nt) {
            v16h b = *(const v16h*)(bFrag + ((size_t)(nt * 7 + kc) * 32 + lane) * 16);
            acc[nt] = __builtin_amdgcn_wmma_f32_16x16x32_f16(
                false, a, false, b, (short)0, acc[nt], false, false);
        }
    }
    // D layout: M = batch_local (r + 8*hi), N = output pixel (nt*16 + row)
#pragma unroll
    for (int nt = 0; nt < 4; ++nt) {
        int n = nt * 16 + row;
        if (n < STEPS) {
#pragma unroll
            for (int r = 0; r < 8; ++r) ldsXp[wv][r + 8 * hi][n] = acc[nt][r];
        }
    }
    __syncthreads();

    // ---------------- Phase 2: 49-step DFR scan over 6 chained cells --------
    v16h Acell[NUM_DFR];
#pragma unroll
    for (int i = 0; i < NUM_DFR; ++i)
        Acell[i] = *(const v16h*)(aFrag + ((size_t)i * 32 + lane) * 16);

    v8f h[NUM_DFR];
#pragma unroll
    for (int i = 0; i < NUM_DFR; ++i)
#pragma unroll
        for (int r = 0; r < 8; ++r) {
            int m = r + 8 * hi;
            h[i][r] = (m < NHID)
                ? cell_out[((size_t)i * Btot + base + row) * NHID + m]
                : 0.0f;
        }

    for (int t = 0; t < STEPS; ++t) {
        float px = ldsXp[wv][row][t];
        // cell 0: D = W0 * px + 0.5*roll(h0), then tanh
        {
            v8f c = roll_feedback(h[0], hi);
            v16h b;
#pragma unroll
            for (int j = 0; j < 16; ++j) b[j] = (_Float16)0.0f;
            if (!hi) b[0] = (_Float16)px;          // K=0 carries the pixel
            v8f d = __builtin_amdgcn_wmma_f32_16x16x32_f16(
                false, Acell[0], false, b, (short)0, c, false, false);
#pragma unroll
            for (int r = 0; r < 8; ++r) h[0][r] = DEV_TANH(d[r]);
        }
        // cells 1..5: D = Wrest[i-1] @ h[i-1] + 0.5*roll(h[i]), then tanh
#pragma unroll
        for (int i = 1; i < NUM_DFR; ++i) {
            v8f  c = roll_feedback(h[i], hi);
            v16h b = pack_state_B(h[i - 1], hi);
            v8f  d = __builtin_amdgcn_wmma_f32_16x16x32_f16(
                false, Acell[i], false, b, (short)0, c, false, false);
#pragma unroll
            for (int r = 0; r < 8; ++r) h[i][r] = DEV_TANH(d[r]);
        }
    }

    // ---------------- Phase 3: head on lanes 0-15 ---------------------------
    float h8 = __shfl_xor(h[NUM_DFR - 1][0], 16, 32);
    float h9 = __shfl_xor(h[NUM_DFR - 1][1], 16, 32);
    if (!hi) {
        float hv[NHID];
#pragma unroll
        for (int r = 0; r < 8; ++r) hv[r] = h[NUM_DFR - 1][r];
        hv[8] = h8; hv[9] = h9;
        float aout[8];
#pragma unroll
        for (int j = 0; j < 8; ++j) {
            float s = 0.0f;
#pragma unroll
            for (int m = 0; m < NHID; ++m) s += ldsW1[j * NHID + m] * hv[m];
            aout[j] = fmaxf(s, 0.0f);
        }
        float* orow = out + (size_t)(base + row) * 10;
#pragma unroll
        for (int c = 0; c < 10; ++c) {
            float s = ldsB2[c];
#pragma unroll
            for (int j = 0; j < 8; ++j) s += ldsW2[c * 8 + j] * aout[j];
            orow[c] = s;
        }
    }
}

// ---------------------------------------------------------------------------
extern "C" void kernel_launch(void* const* d_in, const int* in_sizes, int n_in,
                              void* d_out, int out_size, void* d_ws, size_t ws_size,
                              hipStream_t stream)
{
    const float* x     = (const float*)d_in[0];  // [B,1,14,14]
    const float* cell  = (const float*)d_in[1];  // [6,B,10]
    const float* W_p1  = (const float*)d_in[2];  // [49,196]
    const float* W0    = (const float*)d_in[3];  // [10,1]
    const float* Wrest = (const float*)d_in[4];  // [5,10,10]
    const float* W1    = (const float*)d_in[5];  // [8,10]
    const float* W2    = (const float*)d_in[6];  // [10,8]
    const float* b2    = (const float*)d_in[7];  // [10]
    float* out = (float*)d_out;

    int Btot = in_sizes[0] / KIN;                // 131072

    _Float16* aFrag = (_Float16*)d_ws;           // 6*32*16 halves
    _Float16* bFrag = aFrag + 6 * 32 * 16;       // 28*32*16 halves

    pack_weights_kernel<<<1, 32, 0, stream>>>(W_p1, W0, Wrest, aFrag, bFrag);

    int tiles  = Btot / 16;                      // 8192 (exact)
    int blocks = tiles / 4;                      // 4 waves (tiles) per block
    dfr_fused_kernel<<<blocks, 128, 0, stream>>>(x, cell, W1, W2, b2,
                                                 aFrag, bFrag, out, Btot);
}
